// PINN_DiffReact_85478439125428
// MI455X (gfx1250) — matrix operand order repaired and checked by reading, not compiled, not run on hardware
//
#include <hip/hip_runtime.h>
#include <hip/hip_bf16.h>

typedef __attribute__((ext_vector_type(16))) __bf16 v16bf;
typedef __attribute__((ext_vector_type(8)))  float  v8f;

#define KSTR 264   // 256 + 8 bf16 pad -> row stride 528 B = 132 dwords (== 4 mod 64 banks)
#define PPB  32    // points per workgroup

union FragU { v16bf v; uint4 q[2]; };

__device__ __forceinline__ v16bf ld_frag(const __bf16* p0, const __bf16* p1) {
    FragU u;
    u.q[0] = *(const uint4*)p0;   // elements 0..7   (VGPR 0-3)
    u.q[1] = *(const uint4*)p1;   // elements 8..15  (VGPR 4-7)
    return u.v;
}

// branch-free tanh: 1 - 2/(e^{2z}+1), via hardware v_exp_f32 / v_rcp_f32
__device__ __forceinline__ float fast_tanh(float z) {
    float e = __builtin_amdgcn_exp2f(z * 2.88539008177792681f);   // exp(2z)
    return 1.f - 2.f * __builtin_amdgcn_rcpf(e + 1.f);
}

// Convert W1,W2,W3 (fp32, [k][n] row-major 256x256) -> bf16 W^T [layer][n][k]
__global__ __launch_bounds__(256) void pinn_prep(const float* __restrict__ W1,
                                                 const float* __restrict__ W2,
                                                 const float* __restrict__ W3,
                                                 __bf16* __restrict__ WT) {
    int e = blockIdx.x * 256 + threadIdx.x;       // 3*65536 total
    int m = e >> 16;
    int idx = e & 65535;
    int n = idx >> 8, k = idx & 255;
    const float* W = (m == 0) ? W1 : (m == 1) ? W2 : W3;
    WT[e] = (__bf16)W[k * 256 + n];
}

__global__ __launch_bounds__(512) void pinn_main(
    const float* __restrict__ t, const float* __restrict__ x, const float* __restrict__ y,
    const float* __restrict__ W0, const float* __restrict__ b0,
    const float* __restrict__ b1, const float* __restrict__ b2, const float* __restrict__ b3,
    const float* __restrict__ W4, const float* __restrict__ b4,
    const __bf16* __restrict__ WT,    // [3][256][256] bf16, W^T per middle layer
    float* __restrict__ out, int npts)
{
    // channels: 0=val 1=d/dt 2=d/dx 3=d/dy 4=d2/dx2 5=d2/dy2
    __shared__ __bf16 act[6][PPB][KSTR];     // ~99 KB
    __shared__ float  red[PPB][6][2];

    const int tid  = threadIdx.x;
    const int lane = tid & 31;
    const int wid  = tid >> 5;          // 16 waves
    const int half = lane >> 4;         // 0: lanes 0-15, 1: lanes 16-31
    const int l16  = lane & 15;
    const int p0   = blockIdx.x * PPB;  // 32 points per workgroup

    // wave tiling: rw = point-parity (rows rw*16..rw*16+15 of each channel),
    //              cw = column pair (N-tiles 2cw, 2cw+1)
    const int rw   = wid & 1;
    const int cw   = wid >> 1;
    const int n0a  = cw * 32;
    const int n0b  = cw * 32 + 16;

    // ---------------- layer 0: 3 -> 256 (VALU; seeds all 6 channels) --------
    {
        int p = tid >> 4, i = tid & 15;      // 32 points x 16 column groups
        int g = p0 + p;
        float tv = 0.f, xv = 0.f, yv = 0.f;
        if (g < npts) { tv = t[g]; xv = x[g]; yv = y[g]; }
        #pragma unroll
        for (int jj = 0; jj < 16; ++jj) {
            int n = i * 16 + jj;
            float wt0 = W0[n], wx0 = W0[256 + n], wy0 = W0[512 + n];
            float z  = tv * wt0 + xv * wx0 + yv * wy0 + b0[n];
            float h  = fast_tanh(z);
            float s  = 1.f - h * h;
            act[0][p][n] = (__bf16)h;
            act[1][p][n] = (__bf16)(s * wt0);
            act[2][p][n] = (__bf16)(s * wx0);
            act[3][p][n] = (__bf16)(s * wy0);
            act[4][p][n] = (__bf16)(-2.f * h * s * wx0 * wx0);  // z_xx = 0 at input
            act[5][p][n] = (__bf16)(-2.f * h * s * wy0 * wy0);
        }
    }
    __syncthreads();

    // ---------------- layers 1..3: 256 -> 256 tanh, WMMA -------------------
    #pragma unroll 1
    for (int layer = 0; layer < 3; ++layer) {
        const __bf16* wt = WT + layer * 65536;
        const float*  bp = (layer == 0) ? b1 : (layer == 1) ? b2 : b3;

        // acc[c] = channel-c accumulator for this wave's point-half, 2 N-tiles
        v8f acc[6][2] = {};

        #pragma unroll 1
        for (int k0 = 0; k0 < 256; k0 += 32) {
            // B fragments: column N = n0+l16; lanes 0-15 hold K=k0..k0+15,
            // lanes 16-31 hold K=k0+16..k0+31 (contiguous in W^T rows)
            int kb = k0 + half * 16;
            const __bf16* wb0 = wt + (n0a + l16) * 256 + kb;
            const __bf16* wb1 = wt + (n0b + l16) * 256 + kb;
            v16bf bf0 = ld_frag(wb0, wb0 + 8);
            v16bf bf1 = ld_frag(wb1, wb1 + 8);
            // A fragments: row M = l16 within tile; lanes 0-15: K=k0..k0+15,
            // lanes 16-31: K=k0+16..k0+31
            int ka = k0 + half * 16;
            #pragma unroll
            for (int c = 0; c < 6; ++c) {
                const __bf16* ap = &act[c][rw * 16 + l16][ka];
                v16bf af = ld_frag(ap, ap + 8);
                acc[c][0] = __builtin_amdgcn_wmma_f32_16x16x32_bf16(
                    false, af, false, bf0, (short)0, acc[c][0], false, false);
                acc[c][1] = __builtin_amdgcn_wmma_f32_16x16x32_bf16(
                    false, af, false, bf1, (short)0, acc[c][1], false, false);
            }
        }
        __syncthreads();   // all waves done reading act before in-place update

        #pragma unroll
        for (int j = 0; j < 2; ++j) {
            int n = (j ? n0b : n0a) + l16;
            float bn = bp[n];
            #pragma unroll
            for (int r = 0; r < 8; ++r) {
                int p = rw * 16 + r + half * 8;   // C layout: VGPR r, lanes>=16 -> +8
                float z   = acc[0][j][r] + bn;
                float h   = fast_tanh(z);
                float s   = 1.f - h * h;
                float zt  = acc[1][j][r], zx  = acc[2][j][r], zy  = acc[3][j][r];
                float zxx = acc[4][j][r], zyy = acc[5][j][r];
                float m2hs = -2.f * h * s;
                act[0][p][n] = (__bf16)h;
                act[1][p][n] = (__bf16)(s * zt);
                act[2][p][n] = (__bf16)(s * zx);
                act[3][p][n] = (__bf16)(s * zy);
                act[4][p][n] = (__bf16)(s * zxx + m2hs * zx * zx);
                act[5][p][n] = (__bf16)(s * zyy + m2hs * zy * zy);
            }
        }
        __syncthreads();
    }

    // ---------------- layer 4: 256 -> 2 linear (VALU dot products) ----------
    if (tid < 192) {                 // 32 points x 6 channels, both outputs/thread
        int p = tid & 31;
        int c = tid >> 5;
        const float4* W4q = (const float4*)W4;   // entry q: {k=2q:o0, 2q:o1, 2q+1:o0, 2q+1:o1}
        float s0 = 0.f, s1 = 0.f;
        #pragma unroll 1
        for (int k0 = 0; k0 < 256; k0 += 8) {
            uint4 qa = *(const uint4*)&act[c][p][k0];
            const __bf16* e = (const __bf16*)&qa;
            #pragma unroll
            for (int q = 0; q < 4; ++q) {
                float4 w = W4q[(k0 >> 1) + q];
                float a0 = (float)e[2 * q], a1 = (float)e[2 * q + 1];
                s0 += a0 * w.x + a1 * w.z;
                s1 += a0 * w.y + a1 * w.w;
            }
        }
        if (c == 0) { s0 += b4[0]; s1 += b4[1]; }
        red[p][c][0] = s0;
        red[p][c][1] = s1;
    }
    __syncthreads();

    if (tid < PPB) {
        int g = p0 + tid;
        if (g < npts) {
            float u   = red[tid][0][0], v   = red[tid][0][1];
            float ut  = red[tid][1][0], vt  = red[tid][1][1];
            float uxx = red[tid][4][0], uyy = red[tid][5][0];
            float vxx = red[tid][4][1], vyy = red[tid][5][1];
            float f = u - u * u * u - 0.005f - v + 0.001f * (uxx + uyy) - ut;
            float gg = u - v + 0.005f * (vxx + vyy) - vt;
            out[g]            = u;
            out[npts + g]     = v;
            out[2 * npts + g] = f;
            out[3 * npts + g] = gg;
        }
    }
}

extern "C" void kernel_launch(void* const* d_in, const int* in_sizes, int n_in,
                              void* d_out, int out_size, void* d_ws, size_t ws_size,
                              hipStream_t stream) {
    // inputs: 0=t 1=x 2=y 3=W0 4=b0 5=W1 6=b1 7=W2 8=b2 9=W3 10=b3 11=W4 12=b4
    const float* t  = (const float*)d_in[0];
    const float* x  = (const float*)d_in[1];
    const float* y  = (const float*)d_in[2];
    const float* W0 = (const float*)d_in[3];
    const float* b0 = (const float*)d_in[4];
    const float* W1 = (const float*)d_in[5];
    const float* b1 = (const float*)d_in[6];
    const float* W2 = (const float*)d_in[7];
    const float* b2 = (const float*)d_in[8];
    const float* W3 = (const float*)d_in[9];
    const float* b3 = (const float*)d_in[10];
    const float* W4 = (const float*)d_in[11];
    const float* b4 = (const float*)d_in[12];

    int npts = in_sizes[0];
    __bf16* WT = (__bf16*)d_ws;    // 3 * 256 * 256 * 2 B = 384 KB

    pinn_prep<<<768, 256, 0, stream>>>(W1, W2, W3, WT);

    int nblk = (npts + PPB - 1) / PPB;
    pinn_main<<<nblk, 512, 0, stream>>>(t, x, y, W0, b0, b1, b2, b3, W4, b4, WT,
                                        (float*)d_out, npts);
}
